// ClefDecoder_42812234006557
// MI455X (gfx1250) — matrix operand order repaired and checked by reading, not compiled, not run on hardware
//
#include <hip/hip_runtime.h>
#include <hip/hip_bf16.h>

#define B_       4
#define S_       4096
#define D_       512
#define H_       8
#define DH_      64
#define NB_      128
#define STRIDE_  32
#define WL_      1024
#define LVLOFF_  6144
#define T_       8064
#define NEG_INF_ (-1000000000.0f)

typedef __attribute__((ext_vector_type(16))) __bf16         v16bf;
typedef __attribute__((ext_vector_type(8)))  float          v8f;
typedef __attribute__((ext_vector_type(8)))  unsigned short v8us;

__device__ __forceinline__ unsigned short f2bf(float f) {
  unsigned u = __float_as_uint(f);
  unsigned r = 0x7FFFu + ((u >> 16) & 1u);          // round-to-nearest-even
  return (unsigned short)((u + r) >> 16);
}
__device__ __forceinline__ float bf2f(unsigned short u) {
  return __uint_as_float(((unsigned)u) << 16);
}

// sinusoidal PE: d<256 -> sin(pos * 10000^(-d/256)), else cos with (d-256)
__device__ __forceinline__ float sinpe(float pos, int d) {
  int k = (d < D_ / 2) ? d : d - D_ / 2;
  float inv = expf(-(float)k * (9.210340371976184f / (float)(D_ / 2))); // 10000^(-k/256)
  float ang = pos * inv;
  return (d < D_ / 2) ? sinf(ang) : cosf(ang);
}

// ---------------- prep kernels ----------------
__global__ void k_wtrans(const float* __restrict__ W, unsigned short* __restrict__ Wt) {
  int idx = blockIdx.x * blockDim.x + threadIdx.x;   // D_*D_
  if (idx >= D_ * D_) return;
  int n = idx / D_, k = idx % D_;
  Wt[n * D_ + k] = f2bf(W[k * D_ + n]);              // [K x N] -> [N x K] bf16
}

__global__ void k_ype(const float* __restrict__ y, unsigned short* __restrict__ ype) {
  long idx = (long)blockIdx.x * blockDim.x + threadIdx.x;  // B*S*D
  if (idx >= (long)B_ * S_ * D_) return;
  int d = (int)(idx % D_);
  int j = (int)((idx / D_) % S_);
  float bi = (float)(j / STRIDE_ + 1);
  ype[idx] = f2bf(y[idx] + sinpe(bi, d));
}

__global__ void k_qtok(const unsigned short* __restrict__ ype, unsigned short* __restrict__ qtok) {
  int idx = blockIdx.x * blockDim.x + threadIdx.x;   // B*NB*D
  if (idx >= B_ * NB_ * D_) return;
  int d = idx % D_;
  int i = (idx / D_) % NB_;
  int b = idx / (D_ * NB_);
  qtok[idx] = ype[((long)b * S_ + (long)i * STRIDE_) * D_ + d];
}

__global__ void k_memslice(const float* __restrict__ memory, unsigned short* __restrict__ memb) {
  long idx = (long)blockIdx.x * blockDim.x + threadIdx.x;  // B*WL*D
  if (idx >= (long)B_ * WL_ * D_) return;
  int d = (int)(idx % D_);
  int t = (int)((idx / D_) % WL_);
  int b = (int)(idx / ((long)D_ * WL_));
  memb[idx] = f2bf(memory[((long)b * T_ + LVLOFF_ + t) * (long)D_ + d]);
}

// ---------------- bf16 WMMA GEMM:  C[MxN] = A[MxK] @ Bt[NxK]^T ----------------
// one wave per 16x16 tile, 4 waves/block; f32 accumulate via v_wmma_f32_16x16x32_bf16
__global__ void k_gemm_bf16(const unsigned short* __restrict__ A,
                            const unsigned short* __restrict__ Bt,
                            float* __restrict__ Cf, unsigned short* __restrict__ Cb,
                            int M, int N, int K, int epi) {
  int lane = threadIdx.x & 31;
  int wave = threadIdx.x >> 5;
  int n0 = (blockIdx.x * 4 + wave) * 16;
  int m0 = blockIdx.y * 16;
  int half = lane >> 4;                 // 0: lanes 0-15, 1: lanes 16-31
  int l16  = lane & 15;

  const unsigned short* arow = A  + (long)(m0 + l16) * K;  // A: M=lane%16 (ISA 16-bit A layout)
  const unsigned short* brow = Bt + (long)(n0 + l16) * K;  // B: N=lane%16 (ISA 16-bit B layout)
  int aklo = half ? 8 : 0;              // A: lanes 0-15 hold K 0-7 & 16-23; 16-31 hold 8-15 & 24-31
  int akhi = aklo + 16;
  int bk0  = half ? 8 : 0;              // B: lanes 0-15 hold K 0-7 & 16-23; 16-31 hold 8-15 & 24-31

  v8f c = {};
  for (int kb = 0; kb < K; kb += 32) {
    v8us alo = *(const v8us*)(arow + kb + aklo);
    v8us ahi = *(const v8us*)(arow + kb + akhi);
    v8us blo = *(const v8us*)(brow + kb + bk0);
    v8us bhi = *(const v8us*)(brow + kb + bk0 + 16);
    v16bf av, bv;
#pragma unroll
    for (int i = 0; i < 8; ++i) {
      av[i]     = __builtin_bit_cast(__bf16, (unsigned short)alo[i]);
      av[i + 8] = __builtin_bit_cast(__bf16, (unsigned short)ahi[i]);
      bv[i]     = __builtin_bit_cast(__bf16, (unsigned short)blo[i]);
      bv[i + 8] = __builtin_bit_cast(__bf16, (unsigned short)bhi[i]);
    }
    if (kb + 32 < K) {
      __builtin_prefetch(arow + kb + 32, 0, 1);   // global_prefetch_b8
      __builtin_prefetch(brow + kb + 32, 0, 1);
    }
    c = __builtin_amdgcn_wmma_f32_16x16x32_bf16(false, av, false, bv,
                                                (short)0, c, false, false);
  }

  int col = n0 + l16;
#pragma unroll
  for (int r = 0; r < 8; ++r) {
    int row = m0 + r + 8 * half;        // ISA 32-bit C/D layout
    float v = c[r];
    if (epi == 1) {                     // Kmem epilogue: + time positional encoding
      int t = row & (WL_ - 1);
      float pos = (float)t * ((float)WL_ / (float)(WL_ - 1));
      v += sinpe(pos, col);
    }
    long o = (long)row * N + col;
    if (Cf) Cf[o] = v;
    if (Cb) Cb[o] = f2bf(v);
  }
}

// ---------------- attention 1: bar query over its 31 notes (one wave per b,i,h) ----------------
__global__ void k_attn1(const float* __restrict__ Q,
                        const unsigned short* __restrict__ Kn,
                        const unsigned short* __restrict__ Vn,
                        unsigned short* __restrict__ ctxb) {
  int x = blockIdx.x;
  int h = x % H_;
  int i = (x / H_) % NB_;
  int b = x / (H_ * NB_);
  int lane = threadIdx.x;               // 32
  const float* q = Q + ((long)(b * NB_ + i)) * D_ + h * DH_;

  float s = NEG_INF_;
  if (lane < 31) {                      // tokens 32i+1 .. 32i+31 (never the bar token)
    int tok = i * STRIDE_ + 1 + lane;
    const unsigned short* kp = Kn + ((long)b * S_ + tok) * D_ + h * DH_;
    float acc = 0.f;
#pragma unroll 8
    for (int d = 0; d < DH_; ++d) acc += q[d] * bf2f(kp[d]);
    s = acc * 0.125f;                   // 1/sqrt(64)
  }
  float m = s;
  for (int off = 16; off; off >>= 1) m = fmaxf(m, __shfl_xor(m, off));
  float p = expf(s - m);
  if (lane >= 31) p = 0.f;
  float sum = p;
  for (int off = 16; off; off >>= 1) sum += __shfl_xor(sum, off);
  float pn = p / sum;

  float a0 = 0.f, a1 = 0.f;             // lane accumulates dims 2*lane, 2*lane+1
  for (int j = 0; j < 31; ++j) {
    float pj = __shfl(pn, j);
    const unsigned short* vp = Vn + ((long)b * S_ + (i * STRIDE_ + 1 + j)) * D_ + h * DH_;
    a0 += pj * bf2f(vp[2 * lane]);
    a1 += pj * bf2f(vp[2 * lane + 1]);
  }
  unsigned short* cp = ctxb + ((long)(b * NB_ + i)) * D_ + h * DH_;
  cp[2 * lane]     = f2bf(a0);
  cp[2 * lane + 1] = f2bf(a1);
}

// ---------------- attention 2: 128 queries x 1024 memory keys, com over time ----------------
__global__ void k_attn2(const float* __restrict__ Q2,
                        const unsigned short* __restrict__ Kmem,
                        float* __restrict__ com) {
  __shared__ float qs[H_ * DH_];
  __shared__ float part[H_];
  int bn = blockIdx.x;                  // b*NB_+n
  int b  = bn / NB_;
  int lane = threadIdx.x & 31;
  int h    = threadIdx.x >> 5;          // wave = head
  qs[h * DH_ + lane]      = Q2[(long)bn * D_ + h * DH_ + lane];
  qs[h * DH_ + lane + 32] = Q2[(long)bn * D_ + h * DH_ + lane + 32];
  __syncthreads();

  const float* q = qs + h * DH_;
  const unsigned short* kb = Kmem + (long)b * WL_ * D_ + h * DH_;
  float m = -3.4e38f, s = 0.f, w = 0.f; // online softmax: sum(exp), sum(exp*time_norm)
  for (int it = 0; it < WL_ / 32; ++it) {
    int t = it * 32 + lane;
    const unsigned short* kp = kb + (long)t * D_;
    float acc = 0.f;
#pragma unroll 8
    for (int d = 0; d < DH_; ++d) acc += q[d] * bf2f(kp[d]);
    float sc = acc * 0.125f;
    float nm = fmaxf(m, sc);
    float cf = expf(m - nm);
    float e  = expf(sc - nm);
    s = s * cf + e;
    w = w * cf + e * ((float)t / (float)(WL_ - 1));
    m = nm;
  }
  for (int off = 16; off; off >>= 1) {
    float m2 = __shfl_xor(m, off);
    float s2 = __shfl_xor(s, off);
    float w2 = __shfl_xor(w, off);
    float nm = fmaxf(m, m2);
    float c1 = expf(m - nm), c2 = expf(m2 - nm);
    s = s * c1 + s2 * c2;
    w = w * c1 + w2 * c2;
    m = nm;
  }
  if (lane == 0) part[h] = w / s;
  __syncthreads();
  if (threadIdx.x == 0) {
    float a = 0.f;
    for (int hh = 0; hh < H_; ++hh) a += part[hh];
    com[bn] = a / (float)H_;            // mean over heads
  }
}

// ---------------- outputs ----------------
__global__ void k_out1(const float* __restrict__ com, float* __restrict__ out1) {
  int idx = blockIdx.x * blockDim.x + threadIdx.x;   // B*S
  if (idx >= B_ * S_) return;
  int j = idx % S_, b = idx / S_;
  int seg = j / STRIDE_;
  float v = 0.f;
  if ((j % STRIDE_) == 0 && seg >= 1) v = com[b * NB_ + seg - 1];  // bar i gets com[i-1]
  out1[idx] = v;
}

__global__ void k_out2(const float* __restrict__ summary, float* __restrict__ out2) {
  long idx = (long)blockIdx.x * blockDim.x + threadIdx.x;  // B*S*D
  if (idx >= (long)B_ * S_ * D_) return;
  int d = (int)(idx % D_);
  long jj = idx / D_;
  int j = (int)(jj % S_);
  int b = (int)(jj / S_);
  int seg = j / STRIDE_;
  out2[idx] = (seg == 0) ? 0.f : summary[((long)(b * NB_ + seg - 1)) * D_ + d];
}

extern "C" void kernel_launch(void* const* d_in, const int* in_sizes, int n_in,
                              void* d_out, int out_size, void* d_ws, size_t ws_size,
                              hipStream_t stream) {
  (void)in_sizes; (void)n_in; (void)out_size; (void)ws_size;
  const float* y   = (const float*)d_in[0];
  const float* mem = (const float*)d_in[1];
  const float* Wq  = (const float*)d_in[2];
  const float* Wk  = (const float*)d_in[3];
  const float* Wv  = (const float*)d_in[4];
  const float* Wo  = (const float*)d_in[5];
  const float* Wq2 = (const float*)d_in[6];
  const float* Wkm = (const float*)d_in[7];
  float* out1 = (float*)d_out;
  float* out2 = out1 + (size_t)B_ * S_;

  // workspace layout (ushort elements)
  const size_t E_YPE = (size_t)B_ * S_ * D_;   // 8,388,608
  const size_t E_W   = (size_t)D_ * D_;        // 262,144
  const size_t E_Q   = (size_t)B_ * NB_ * D_;  // 262,144
  unsigned short* ws_us = (unsigned short*)d_ws;
  size_t off = 0;
  unsigned short* ype  = ws_us + off; off += E_YPE;
  unsigned short* Knb  = ws_us + off; off += E_YPE;
  unsigned short* Vnb  = ws_us + off; off += E_YPE;
  unsigned short* WT   = ws_us + off; off += 6 * E_W;   // WqT,WkT,WvT,WoT,Wq2T,WkmT
  unsigned short* qtok = ws_us + off; off += E_Q;
  unsigned short* ctxb = ws_us + off; off += E_Q;
  unsigned short* sumb = ws_us + off; off += E_Q;
  float* Qf   = (float*)(ws_us + off);
  float* sumf = Qf   + E_Q;
  float* Q2f  = sumf + E_Q;
  float* com  = Q2f  + E_Q;
  // aliases (lifetimes disjoint on the in-order stream)
  unsigned short* memb  = ype;   // B*WL*D <= B*S*D
  unsigned short* Kmemb = Knb;

  const int TPB = 256;
  k_wtrans<<<(D_ * D_ + TPB - 1) / TPB, TPB, 0, stream>>>(Wq,  WT + 0 * E_W);
  k_wtrans<<<(D_ * D_ + TPB - 1) / TPB, TPB, 0, stream>>>(Wk,  WT + 1 * E_W);
  k_wtrans<<<(D_ * D_ + TPB - 1) / TPB, TPB, 0, stream>>>(Wv,  WT + 2 * E_W);
  k_wtrans<<<(D_ * D_ + TPB - 1) / TPB, TPB, 0, stream>>>(Wo,  WT + 3 * E_W);
  k_wtrans<<<(D_ * D_ + TPB - 1) / TPB, TPB, 0, stream>>>(Wq2, WT + 4 * E_W);
  k_wtrans<<<(D_ * D_ + TPB - 1) / TPB, TPB, 0, stream>>>(Wkm, WT + 5 * E_W);

  k_ype<<<(unsigned)(E_YPE / TPB), TPB, 0, stream>>>(y, ype);
  k_qtok<<<(unsigned)(E_Q / TPB), TPB, 0, stream>>>(ype, qtok);

  dim3 gK(D_ / 64, (B_ * S_) / 16);          // 8 x 1024 tiles
  k_gemm_bf16<<<gK, 128, 0, stream>>>(ype, WT + 1 * E_W, nullptr, Knb, B_ * S_, D_, D_, 0);
  k_gemm_bf16<<<gK, 128, 0, stream>>>(ype, WT + 2 * E_W, nullptr, Vnb, B_ * S_, D_, D_, 0);

  dim3 gQ(D_ / 64, (B_ * NB_) / 16);         // 8 x 32
  k_gemm_bf16<<<gQ, 128, 0, stream>>>(qtok, WT + 0 * E_W, Qf, nullptr, B_ * NB_, D_, D_, 0);

  k_attn1<<<B_ * NB_ * H_, 32, 0, stream>>>(Qf, Knb, Vnb, ctxb);

  k_gemm_bf16<<<gQ, 128, 0, stream>>>(ctxb, WT + 3 * E_W, sumf, sumb, B_ * NB_, D_, D_, 0);
  k_gemm_bf16<<<gQ, 128, 0, stream>>>(sumb, WT + 4 * E_W, Q2f, nullptr, B_ * NB_, D_, D_, 0);

  k_memslice<<<(unsigned)(((size_t)B_ * WL_ * D_) / TPB), TPB, 0, stream>>>(mem, memb);
  dim3 gM(D_ / 64, (B_ * WL_) / 16);         // 8 x 256
  k_gemm_bf16<<<gM, 128, 0, stream>>>(memb, WT + 5 * E_W, nullptr, Kmemb, B_ * WL_, D_, D_, 1);

  k_attn2<<<B_ * NB_, 256, 0, stream>>>(Q2f, Kmemb, com);

  k_out1<<<(B_ * S_ + TPB - 1) / TPB, TPB, 0, stream>>>(com, out1);
  k_out2<<<(unsigned)(E_YPE / TPB), TPB, 0, stream>>>(sumf, out2);
}
